// MultiHeadAttention_6210522710499
// MI455X (gfx1250) — matrix operand order repaired and checked by reading, not compiled, not run on hardware
//
#include <hip/hip_runtime.h>

typedef __attribute__((ext_vector_type(16))) _Float16 v16h;
typedef __attribute__((ext_vector_type(8)))  _Float16 v8h;
typedef __attribute__((ext_vector_type(4)))  _Float16 v4h;
typedef __attribute__((ext_vector_type(8)))  float    v8f;

#define HD   64
#define NH   16
#define SEQ  2048
#define DM   1024
#define BATCH 4
#define ROWS (BATCH * SEQ)   // 8192

// ---------------- WMMA fragment helpers (per CDNA5 ISA layout tables) ----------------

// A-matrix 16x32 f16: lane<16 -> row=lane, halves {K..K+7, K+16..K+23};
//                     lane>=16 -> same row, halves {K+8..K+15, K+24..K+31}
__device__ __forceinline__ v16h load_a16(const _Float16* p, int lane) {
    const int o = (lane & 16) ? 8 : 0;
    union { v16h v; v8h h[2]; } u;
    u.h[0] = *(const v8h*)(p + o);
    u.h[1] = *(const v8h*)(p + o + 16);
    return u.v;
}

// B-matrix 32x16 f16 (column per lane, K-contiguous source):
//   lane<16 -> K 0..7, 16..23; lane>=16 -> K 8..15, 24..31
__device__ __forceinline__ v16h load_b16(const _Float16* p, int lane) {
    const int o = (lane & 16) ? 8 : 0;
    union { v16h v; v8h h[2]; } u;
    u.h[0] = *(const v8h*)(p + o);
    u.h[1] = *(const v8h*)(p + o + 16);
    return u.v;
}

__device__ __forceinline__ v8f wmma_f16(v16h a, v16h b, v8f c) {
    return __builtin_amdgcn_wmma_f32_16x16x32_f16(false, a, false, b,
                                                  (short)0, c, false, false);
}

// ---------------- 1) fp32 -> fp16 convert (x) ----------------
__global__ void cvt_f32_f16(const float* __restrict__ x, _Float16* __restrict__ y) {
    size_t i = (size_t)blockIdx.x * blockDim.x + threadIdx.x;
    float4 v = ((const float4*)x)[i];
    v4h o;
    o.x = (_Float16)v.x; o.y = (_Float16)v.y;
    o.z = (_Float16)v.z; o.w = (_Float16)v.w;
    ((v4h*)y)[i] = o;
}

// ---------------- 2) weight transpose + convert: Wt[n][k] = (f16) W[k][n] ----------------
__global__ void transpose_w_f16(const float* __restrict__ W, _Float16* __restrict__ Wt) {
    __shared__ _Float16 tile[32][33];
    const int tx = threadIdx.x & 31;
    const int ty = threadIdx.x >> 5;          // 0..7
    const int kb = blockIdx.x << 5;
    const int nb = blockIdx.y << 5;
#pragma unroll
    for (int r = 0; r < 4; ++r) {
        int k = kb + ty + r * 8;
        tile[ty + r * 8][tx] = (_Float16)W[(size_t)k * DM + nb + tx];
    }
    __syncthreads();
#pragma unroll
    for (int r = 0; r < 4; ++r) {
        int n = nb + ty + r * 8;
        Wt[(size_t)n * DM + kb + tx] = tile[tx][ty + r * 8];
    }
}

// ---------------- 3) WMMA GEMM, 32x64 tile per wave, ping-pong pipelined ----------------
// C[8192x1024] = A f16 @ Wt^T + bias
// mode 0: f16 out, [B,H,S,HD]        (Q, K)
// mode 1: f16 out, [B,H,HD,S]        (V transposed)
// mode 2: f32 out, [M,N] row-major   (final projection -> d_out)
__global__ void __launch_bounds__(256, 1)
gemm_wmma(const _Float16* __restrict__ A, const _Float16* __restrict__ Bt,
          const float* __restrict__ bias, _Float16* __restrict__ outh,
          float* __restrict__ outf, int mode) {
    const int lane = threadIdx.x & 31;
    const int wave = blockIdx.x * (blockDim.x >> 5) + (threadIdx.x >> 5);
    const int m0 = (wave & 255) << 5;   // 256 tiles of 32 rows over M=8192
    const int n0 = (wave >> 8) << 6;    // 16 groups of 64 over N=1024
    const int lrow = lane & 15;

    v8f acc[8] = {};                    // [m-sub 0..1][n-sub 0..3]
    const _Float16* arow0 = A + (size_t)(m0 + lrow) * DM;
    const _Float16* arow1 = arow0 + (size_t)16 * DM;
    const _Float16* brow[4];
#pragma unroll
    for (int f = 0; f < 4; ++f)
        brow[f] = Bt + (size_t)(n0 + 16 * f + lrow) * DM;

    auto load_set = [&](int kk, v16h& A0, v16h& A1,
                        v16h& B0, v16h& B1, v16h& B2, v16h& B3) {
        A0 = load_a16(arow0 + kk, lane);
        A1 = load_a16(arow1 + kk, lane);
        B0 = load_b16(brow[0] + kk, lane);
        B1 = load_b16(brow[1] + kk, lane);
        B2 = load_b16(brow[2] + kk, lane);
        B3 = load_b16(brow[3] + kk, lane);
    };
    auto do_wmmas = [&](v16h A0, v16h A1,
                        v16h B0, v16h B1, v16h B2, v16h B3) {
        acc[0] = wmma_f16(A0, B0, acc[0]);
        acc[1] = wmma_f16(A0, B1, acc[1]);
        acc[2] = wmma_f16(A0, B2, acc[2]);
        acc[3] = wmma_f16(A0, B3, acc[3]);
        acc[4] = wmma_f16(A1, B0, acc[4]);
        acc[5] = wmma_f16(A1, B1, acc[5]);
        acc[6] = wmma_f16(A1, B2, acc[6]);
        acc[7] = wmma_f16(A1, B3, acc[7]);
    };

    // ping-pong buffers: set X holds k, set Y holds k+32
    v16h a0X, a1X, b0X, b1X, b2X, b3X;
    v16h a0Y, a1Y, b0Y, b1Y, b2Y, b3Y;
    load_set(0,  a0X, a1X, b0X, b1X, b2X, b3X);
    load_set(32, a0Y, a1Y, b0Y, b1Y, b2Y, b3Y);

    // main loop: compute {k, k+32} while reloading the just-consumed buffers
    // with {k+64, k+96}; data steps 0..992, loop covers k = 0..896
#pragma clang loop unroll(disable)
    for (int k = 0; k < DM - 64; k += 64) {
        __builtin_prefetch(arow0 + k + 192, 0, 1);   // global_prefetch_b8
        __builtin_prefetch(arow1 + k + 192, 0, 1);
        do_wmmas(a0X, a1X, b0X, b1X, b2X, b3X);
        load_set(k + 64, a0X, a1X, b0X, b1X, b2X, b3X);
        do_wmmas(a0Y, a1Y, b0Y, b1Y, b2Y, b3Y);
        load_set(k + 96, a0Y, a1Y, b0Y, b1Y, b2Y, b3Y);
    }
    // tail: last two preloaded k-steps (960, 992)
    do_wmmas(a0X, a1X, b0X, b1X, b2X, b3X);
    do_wmmas(a0Y, a1Y, b0Y, b1Y, b2Y, b3Y);

    const int rhalf = (lane & 16) ? 8 : 0;
#pragma unroll
    for (int g = 0; g < 2; ++g) {
#pragma unroll
        for (int f = 0; f < 4; ++f) {
            const int col = n0 + 16 * f + lrow;
            const float bv = bias[col];
#pragma unroll
            for (int i = 0; i < 8; ++i) {
                const int r = m0 + 16 * g + rhalf + i;
                const float val = acc[g * 4 + f][i] + bv;
                if (mode == 0) {
                    int b = r >> 11, s = r & (SEQ - 1);
                    int h = col >> 6, d = col & (HD - 1);
                    outh[(((size_t)(b * NH + h) * SEQ + s) << 6) + d] = (_Float16)val;
                } else if (mode == 1) {
                    int b = r >> 11, s = r & (SEQ - 1);
                    int h = col >> 6, d = col & (HD - 1);
                    outh[(((size_t)(b * NH + h) * HD + d) << 11) + s] = (_Float16)val;
                } else {
                    outf[(size_t)r * DM + col] = val;
                }
            }
        }
    }
}

// ---------------- 4) flash attention: one wave per 16-query tile, pipelined ----------------
__global__ void __launch_bounds__(256, 1)
attn_fwd(const _Float16* __restrict__ Q, const _Float16* __restrict__ Km,
         const _Float16* __restrict__ Vt, _Float16* __restrict__ Oout) {
    __shared__ __align__(16) _Float16 plds[8][16][32];   // P staging, 1KB per wave
    const int lane = threadIdx.x & 31;
    const int w = threadIdx.x >> 5;
    const int wave = blockIdx.x * 8 + w;
    const int qt = wave & 127;                  // 128 q-tiles per (b,h)
    const int bh = wave >> 7;                   // 64 (b,h) pairs
    const int lrow = lane & 15;
    const int hi8 = (lane & 16) ? 8 : 0;

    const _Float16* Qb = Q + ((size_t)bh * SEQ + qt * 16) * HD;
    const _Float16* Kb = Km + (size_t)bh * SEQ * HD;
    const _Float16* Vb = Vt + (size_t)bh * HD * SEQ;

    // Q A-fragments, pre-scaled by 1/sqrt(HD) = 0.125 (exact in f16)
    v16h qa0 = load_a16(Qb + (size_t)lrow * HD, lane) * (_Float16)0.125f;
    v16h qa1 = load_a16(Qb + (size_t)lrow * HD + 32, lane) * (_Float16)0.125f;

    v8f o0 = {}, o1 = {}, o2 = {}, o3 = {};
    float m[8], l[8];
#pragma unroll
    for (int i = 0; i < 8; ++i) { m[i] = -1e30f; l[i] = 0.f; }

    // preload K fragments for first key block
    v16h kf0, kf1, kf2, kf3;
    {
        const _Float16* kr = Kb + (size_t)lrow * HD;
        kf0 = load_b16(kr, lane);
        kf1 = load_b16(kr + 32, lane);
        kf2 = load_b16(kr + 16 * HD, lane);
        kf3 = load_b16(kr + 16 * HD + 32, lane);
    }

#pragma clang loop unroll(disable)
    for (int kb = 0; kb < SEQ; kb += 32) {
        // ---- S = (Q*scale) @ K^T  (two 16x16 key sub-tiles) ----
        v8f s0 = {}, s1 = {};
        s0 = wmma_f16(qa0, kf0, s0);
        s0 = wmma_f16(qa1, kf1, s0);
        s1 = wmma_f16(qa0, kf2, s1);
        s1 = wmma_f16(qa1, kf3, s1);

        // issue V fragment loads now; consumed only after softmax + LDS round-trip
        const _Float16* vr = Vb + (size_t)lrow * SEQ + kb;
        v16h vf0 = load_b16(vr,            lane);
        v16h vf1 = load_b16(vr + 16 * SEQ, lane);
        v16h vf2 = load_b16(vr + 32 * SEQ, lane);
        v16h vf3 = load_b16(vr + 48 * SEQ, lane);

        // ---- online softmax (rows live in 16-lane halves, one row per vgpr) ----
        float t[8], p0[8], p1[8], corr[8];
#pragma unroll
        for (int i = 0; i < 8; ++i) t[i] = fmaxf(s0[i], s1[i]);
#pragma unroll
        for (int msk = 1; msk <= 8; msk <<= 1)
#pragma unroll
            for (int i = 0; i < 8; ++i)
                t[i] = fmaxf(t[i], __shfl_xor(t[i], msk, 16));
#pragma unroll
        for (int i = 0; i < 8; ++i) {
            float mn = fmaxf(m[i], t[i]);
            corr[i] = __expf(m[i] - mn);
            m[i] = mn;
            p0[i] = __expf(s0[i] - mn);
            p1[i] = __expf(s1[i] - mn);
            t[i] = p0[i] + p1[i];
        }
#pragma unroll
        for (int msk = 1; msk <= 8; msk <<= 1)
#pragma unroll
            for (int i = 0; i < 8; ++i)
                t[i] += __shfl_xor(t[i], msk, 16);
#pragma unroll
        for (int i = 0; i < 8; ++i) l[i] = l[i] * corr[i] + t[i];
#pragma unroll
        for (int i = 0; i < 8; ++i) {
            o0[i] *= corr[i]; o1[i] *= corr[i];
            o2[i] *= corr[i]; o3[i] *= corr[i];
        }

        // ---- re-swizzle P: C-layout -> A-fragment via LDS (same-wave DS is in-order) ----
#pragma unroll
        for (int i = 0; i < 8; ++i) {
            plds[w][hi8 + i][lrow]      = (_Float16)p0[i];
            plds[w][hi8 + i][lrow + 16] = (_Float16)p1[i];
        }
        v16h pa = load_a16(&plds[w][lrow][0], lane);

        // preload next key block's K fragments before the PV WMMAs
        if (kb + 32 < SEQ) {
            const _Float16* kr = Kb + (size_t)(kb + 32 + lrow) * HD;
            kf0 = load_b16(kr, lane);
            kf1 = load_b16(kr + 32, lane);
            kf2 = load_b16(kr + 16 * HD, lane);
            kf3 = load_b16(kr + 16 * HD + 32, lane);
        }

        // ---- O += P @ V  (V stored transposed [HD, S] -> contiguous B-frags) ----
        o0 = wmma_f16(pa, vf0, o0);
        o1 = wmma_f16(pa, vf1, o1);
        o2 = wmma_f16(pa, vf2, o2);
        o3 = wmma_f16(pa, vf3, o3);
    }

    // ---- epilogue: divide by row-sum, store f16 [B,S,D] ----
    const int b = bh >> 4, h = bh & 15;
    const int sbase = qt * 16 + hi8;
#pragma unroll
    for (int i = 0; i < 8; ++i) {
        const float inv = 1.0f / l[i];
        const size_t rowoff = ((size_t)(b * SEQ + sbase + i)) * DM + h * HD;
        Oout[rowoff + lrow]      = (_Float16)(o0[i] * inv);
        Oout[rowoff + lrow + 16] = (_Float16)(o1[i] * inv);
        Oout[rowoff + lrow + 32] = (_Float16)(o2[i] * inv);
        Oout[rowoff + lrow + 48] = (_Float16)(o3[i] * inv);
    }
}

// ---------------- host side ----------------
extern "C" void kernel_launch(void* const* d_in, const int* in_sizes, int n_in,
                              void* d_out, int out_size, void* d_ws, size_t ws_size,
                              hipStream_t stream) {
    (void)in_sizes; (void)n_in; (void)out_size; (void)ws_size;
    const float* x  = (const float*)d_in[0];
    const float* wq = (const float*)d_in[1];
    const float* bq = (const float*)d_in[2];
    const float* wk = (const float*)d_in[3];
    const float* bk = (const float*)d_in[4];
    const float* wv = (const float*)d_in[5];
    const float* bv = (const float*)d_in[6];
    const float* wo = (const float*)d_in[7];
    const float* bo = (const float*)d_in[8];
    float* out = (float*)d_out;

    char* ws = (char*)d_ws;
    const size_t MB = (size_t)1 << 20;
    _Float16* x16  = (_Float16*)(ws);             // 16MB; later reused as attention output
    _Float16* q16  = (_Float16*)(ws + 16 * MB);   // 16MB  [B,H,S,HD]
    _Float16* k16  = (_Float16*)(ws + 32 * MB);   // 16MB  [B,H,S,HD]
    _Float16* v16t = (_Float16*)(ws + 48 * MB);   // 16MB  [B,H,HD,S]
    _Float16* wqt  = (_Float16*)(ws + 64 * MB);   // 2MB each, transposed f16 weights
    _Float16* wkt  = (_Float16*)(ws + 66 * MB);
    _Float16* wvt  = (_Float16*)(ws + 68 * MB);
    _Float16* wot  = (_Float16*)(ws + 70 * MB);   // total 72MB

    // x -> f16
    cvt_f32_f16<<<(ROWS * DM / 4) / 256, 256, 0, stream>>>(x, x16);

    // weights -> transposed f16
    dim3 tg(DM / 32, DM / 32);
    transpose_w_f16<<<tg, 256, 0, stream>>>(wq, wqt);
    transpose_w_f16<<<tg, 256, 0, stream>>>(wk, wkt);
    transpose_w_f16<<<tg, 256, 0, stream>>>(wv, wvt);
    transpose_w_f16<<<tg, 256, 0, stream>>>(wo, wot);

    // Q/K/V projections (4096 waves each: 256 M-tiles x 16 N-groups, 8 waves/block)
    gemm_wmma<<<512, 256, 0, stream>>>(x16, wqt, bq, q16, nullptr, 0);
    gemm_wmma<<<512, 256, 0, stream>>>(x16, wkt, bk, k16, nullptr, 0);
    gemm_wmma<<<512, 256, 0, stream>>>(x16, wvt, bv, v16t, nullptr, 1);

    // attention (8192 waves = 64 bh * 128 q-tiles), writes f16 [B,S,D] over x16
    attn_fwd<<<1024, 256, 0, stream>>>(q16, k16, v16t, x16);

    // output projection -> f32 d_out
    gemm_wmma<<<512, 256, 0, stream>>>(x16, wot, bo, nullptr, out, 2);
}
